// DualCrossMessageBlock_40475771797589
// MI455X (gfx1250) — matrix operand order repaired and checked by reading, not compiled, not run on hardware
//
#include <hip/hip_runtime.h>
#include <hip/hip_bf16.h>

typedef __attribute__((ext_vector_type(16))) __bf16 v16bf;
typedef __attribute__((ext_vector_type(8)))  float  v8f;

#define F128 128
#define SIXF 768
#define RDIM 20

__device__ inline v8f wmma_bf16(v16bf a, v16bf b, v8f c) {
  // (neg_a, A, neg_b, B, c_mod, C, reuse_a, reuse_b)
  return __builtin_amdgcn_wmma_f32_16x16x32_bf16(false, a, false, b, (short)0, c, false, false);
}

__device__ inline v16bf ldfrag(const __bf16* p) {
  return *reinterpret_cast<const v16bf*>(p);  // 32B aligned -> global/ds b128 x2
}

// A-fragment addressing (16-bit 16x32): element (m, kk) -> lane, j
//   lane = ((kk>>3)&1)*16 + m ;  j = (kk&7) | ((kk>>1)&8)
// Inverse (pack): given (lane,j): m=lane&15, half=lane>>4,
//   kk = (j&7) + ((j>>3)<<4) + half*8

// ---------- Kernel: out = [s | v] (edges then accumulate ds/dv) ----------
__global__ void init_out_kernel(const float* __restrict__ s, const float* __restrict__ v,
                                float* __restrict__ out, int N) {
  int idx = blockIdx.x * blockDim.x + threadIdx.x;
  int nS = N * F128;
  int total = nS * 4;
  if (idx < nS)         out[idx] = s[idx];
  else if (idx < total) out[idx] = v[idx - nS];
}

// ---------- Pack s (N x 128 f32) into A-fragment bf16 layout ----------
// sp[rt][ks][lane][j], rt = row tile (16 rows), ks = k-step (32 k)
__global__ void pack_s_kernel(const float* __restrict__ s, unsigned short* __restrict__ sp_u16,
                              int N) {
  __bf16* sp = reinterpret_cast<__bf16*>(sp_u16);
  int idx = blockIdx.x * blockDim.x + threadIdx.x;
  if (idx >= N * F128) return;
  int rt   = idx >> 11;           // / 2048
  int rem  = idx & 2047;
  int ks   = rem >> 9;            // / 512
  int lane = (rem >> 4) & 31;
  int j    = rem & 15;
  int m = lane & 15, half = lane >> 4;
  int k = ks * 32 + (j & 7) + ((j >> 3) << 4) + half * 8;
  sp[idx] = (__bf16)s[(size_t)(rt * 16 + m) * F128 + k];
}

// ---------- Pack a row-major f32 weight (K x C) into B-fragment bf16 layout ----
// wp[ct][ks][lane][j]; ct = 16-col tile, ks = 32-k step; k = ks*32 + (j&7) + (j>>3)*16 + (lane>>4)*8
__global__ void pack_w_kernel(const float* __restrict__ W, unsigned short* __restrict__ wp_u16,
                              int nct, int nks, int ldb) {
  __bf16* wp = reinterpret_cast<__bf16*>(wp_u16);
  int idx = blockIdx.x * blockDim.x + threadIdx.x;
  if (idx >= nct * nks * 512) return;
  int t    = idx >> 9;            // ct*nks + ks
  int lane = (idx >> 4) & 31;
  int j    = idx & 15;
  int ct = t / nks, ks = t % nks;
  int n = lane & 15, halfb = lane >> 4;
  int k = ks * 32 + (j & 7) + ((j >> 3) << 4) + halfb * 8;
  int c = ct * 16 + n;
  wp[idx] = (__bf16)W[(size_t)k * ldb + c];
}

// ---------- Pack [Wr ; Wr ; 0] (64 x 768) into B-fragment bf16 layout ----------
__global__ void pack_bpad_kernel(const float* __restrict__ Wr, unsigned short* __restrict__ bp_u16) {
  __bf16* bp = reinterpret_cast<__bf16*>(bp_u16);
  int idx = blockIdx.x * blockDim.x + threadIdx.x;
  if (idx >= 48 * 2 * 512) return;
  int t    = idx >> 9;            // ct*2 + ks
  int lane = (idx >> 4) & 31;
  int j    = idx & 15;
  int ct = t >> 1, ks = t & 1;
  int n = lane & 15, halfb = lane >> 4;
  int k = ks * 32 + (j & 7) + ((j >> 3) << 4) + halfb * 8;
  int c = ct * 16 + n;
  float val = 0.f;
  if (k < RDIM)           val = Wr[k * SIXF + c];
  else if (k < 2 * RDIM)  val = Wr[(k - RDIM) * SIXF + c];
  bp[idx] = (__bf16)val;
}

// ---------- h = silu(s @ W1 + b1), written in A-fragment layout ----------
__global__ void __launch_bounds__(256) node_hidden_kernel(
    const unsigned short* __restrict__ sp_u16, const unsigned short* __restrict__ w1p_u16,
    const float* __restrict__ b1, unsigned short* __restrict__ hp_u16, int N) {
  const __bf16* sp  = reinterpret_cast<const __bf16*>(sp_u16);
  const __bf16* w1p = reinterpret_cast<const __bf16*>(w1p_u16);
  __bf16* hp = reinterpret_cast<__bf16*>(hp_u16);
  int lane = threadIdx.x & 31, wave = threadIdx.x >> 5;
  int rt = blockIdx.x;
  const __bf16* spt = sp + (size_t)rt * 2048;
  v8f c = {};
#pragma unroll
  for (int ks = 0; ks < 4; ++ks) {
    v16bf a = ldfrag(spt + (ks * 32 + lane) * 16);
    v16bf b = ldfrag(w1p + ((wave * 4 + ks) * 32 + lane) * 16);
    c = wmma_bf16(a, b, c);
  }
  int halfd = (lane >> 4) & 1, n = lane & 15;
#pragma unroll
  for (int r = 0; r < 8; ++r) {
    int row = r + halfd * 8;
    int col = wave * 16 + n;
    float x = c[r] + b1[col];
    float y = x / (1.0f + __expf(-x));  // silu
    // store into A-fragment layout for the next GEMM (k = col)
    int ks3 = col >> 5, kk = col & 31;
    int ha = (kk >> 3) & 1, j = (kk & 7) | ((kk >> 1) & 8);
    hp[(size_t)rt * 2048 + (ks3 * 32 + ha * 16 + row) * 16 + j] = (__bf16)y;
  }
}

// ---------- phi = h @ W2 + b2, f32 (N x 768) ----------
__global__ void __launch_bounds__(256) node_phi_kernel(
    const unsigned short* __restrict__ hp_u16, const unsigned short* __restrict__ w2p_u16,
    const float* __restrict__ b2, float* __restrict__ phi, int N) {
  const __bf16* hp  = reinterpret_cast<const __bf16*>(hp_u16);
  const __bf16* w2p = reinterpret_cast<const __bf16*>(w2p_u16);
  int lane = threadIdx.x & 31, wave = threadIdx.x >> 5;
  int rt = blockIdx.x;
  v16bf a[4];
#pragma unroll
  for (int i = 0; i < 4; ++i) a[i] = ldfrag(hp + (size_t)rt * 2048 + (i * 32 + lane) * 16);
  int halfd = (lane >> 4) & 1, n = lane & 15;
#pragma unroll
  for (int t = 0; t < 6; ++t) {     // 8 waves x 6 tiles = 768 cols
    int ct = wave * 6 + t;
    v8f c = {};
#pragma unroll
    for (int i = 0; i < 4; ++i) {
      v16bf b = ldfrag(w2p + ((ct * 4 + i) * 32 + lane) * 16);
      c = wmma_bf16(a[i], b, c);
    }
#pragma unroll
    for (int r = 0; r < 8; ++r) {
      int row = r + halfd * 8, col = ct * 16 + n;
      phi[(size_t)(rt * 16 + row) * SIXF + col] = c[r] + b2[col];
    }
  }
}

// ---------- Fused edge kernel: 16 edges / block ----------
__global__ void __launch_bounds__(256, 1) edge_kernel(
    const float* __restrict__ re1, const float* __restrict__ re2,
    const float* __restrict__ f1, const float* __restrict__ f2,
    const float* __restrict__ u1, const float* __restrict__ u2,
    const int* __restrict__ eidx, const float* __restrict__ phi,
    const float* __restrict__ v, const unsigned short* __restrict__ bp_u16,
    const float* __restrict__ br, float* __restrict__ outS,
    float* __restrict__ outV, int E) {
  const __bf16* bp = reinterpret_cast<const __bf16*>(bp_u16);
  __shared__ __align__(32) __bf16 sAb[2 * 32 * 16];  // A frags (K=64), 2KB
  __shared__ float sW[16 * SIXF];                    // W tile, 48KB
  __shared__ float sF[16];
  __shared__ int   sI[16], sJ[16];
  __shared__ float sU1[48], sU2[48];

  int t = threadIdx.x;
  int eb = blockIdx.x * 16;

  // ---- Phase 1: A = [re1*f1 | re2*f2 | 0] staged directly in frag layout ----
#pragma unroll
  for (int it = 0; it < 4; ++it) {
    int idx = it * 256 + t;            // 16 edges x 64 k
    int e = idx >> 6, k = idx & 63;
    int ed = eb + e;
    float val = 0.f;
    if (k < RDIM)           val = re1[ed * RDIM + k] * f1[ed];
    else if (k < 2 * RDIM)  val = re2[ed * RDIM + (k - RDIM)] * f2[ed];
    int ks = k >> 5, kk = k & 31;
    int ha = (kk >> 3) & 1, j = (kk & 7) | ((kk >> 1) & 8);
    sAb[(ks * 32 + ha * 16 + e) * 16 + j] = (__bf16)val;
  }
  if (t < 16) {
    int ed = eb + t;
    sF[t] = f1[ed] + f2[ed];
    sI[t] = eidx[ed];           // idx_i
    sJ[t] = eidx[E + ed];       // idx_j
  }
  if (t < 48) {
    int e = t / 3, d = t % 3;
    sU1[t] = u1[(eb + e) * 3 + d];
    sU2[t] = u2[(eb + e) * 3 + d];
  }
  __syncthreads();

  // ---- Phase 2: W = A @ [Wr;Wr] + (f1+f2)*br via WMMA ----
  int lane = t & 31, wave = t >> 5;
  v16bf a0 = ldfrag(&sAb[lane * 16]);
  v16bf a1 = ldfrag(&sAb[(32 + lane) * 16]);
  int halfd = (lane >> 4) & 1, n = lane & 15;
#pragma unroll
  for (int i = 0; i < 6; ++i) {
    int ct = wave * 6 + i;
    v8f c = {};
    v16bf b0 = ldfrag(bp + ((ct * 2 + 0) * 32 + lane) * 16);
    c = wmma_bf16(a0, b0, c);
    v16bf b1v = ldfrag(bp + ((ct * 2 + 1) * 32 + lane) * 16);
    c = wmma_bf16(a1, b1v, c);
#pragma unroll
    for (int r = 0; r < 8; ++r) {
      int row = r + halfd * 8, col = ct * 16 + n;
      sW[row * SIXF + col] = c[r] + sF[row] * br[col];
    }
  }
  __syncthreads();

  // ---- Phase 3: x = phi[idx_j] * W, cross combine, atomic scatter ----
#pragma unroll
  for (int it = 0; it < 8; ++it) {
    int idx = it * 256 + t;            // 16 edges x 128 features
    int e = idx >> 7, f = idx & 127;
    int j = sJ[e], ii = sI[e];
    const float* ph = phi + (size_t)j * SIXF;
    const float* w  = sW + e * SIXF;
    float xs   = ph[f]            * w[f];
    float xvv  = ph[F128 + f]     * w[F128 + f];
    float xvs1 = ph[2 * F128 + f] * w[2 * F128 + f];
    float xvs2 = ph[3 * F128 + f] * w[3 * F128 + f];
    float xvc1 = ph[4 * F128 + f] * w[4 * F128 + f];
    float xvc2 = ph[5 * F128 + f] * w[5 * F128 + f];
    const float* vj = v + (size_t)j * 3 * F128;
    float v0 = vj[f], v1 = vj[F128 + f], v2 = vj[2 * F128 + f];
    float ax = sU1[e * 3], ay = sU1[e * 3 + 1], az = sU1[e * 3 + 2];
    float bx = sU2[e * 3], by = sU2[e * 3 + 1], bz = sU2[e * 3 + 2];
    float c1x = v1 * az - v2 * ay, c1y = v2 * ax - v0 * az, c1z = v0 * ay - v1 * ax;
    float c2x = v1 * bz - v2 * by, c2y = v2 * bx - v0 * bz, c2z = v0 * by - v1 * bx;
    float xv0 = v0 * xvv + xvs1 * ax + xvs2 * bx + xvc1 * c1x + xvc2 * c2x;
    float xv1 = v1 * xvv + xvs1 * ay + xvs2 * by + xvc1 * c1y + xvc2 * c2y;
    float xv2 = v2 * xvv + xvs1 * az + xvs2 * bz + xvc1 * c1z + xvc2 * c2z;
    unsafeAtomicAdd(&outS[(size_t)ii * F128 + f], xs);
    float* ov = outV + (size_t)ii * 3 * F128 + f;
    unsafeAtomicAdd(ov,            xv0);
    unsafeAtomicAdd(ov + F128,     xv1);
    unsafeAtomicAdd(ov + 2 * F128, xv2);
  }
}

extern "C" void kernel_launch(void* const* d_in, const int* in_sizes, int n_in,
                              void* d_out, int out_size, void* d_ws, size_t ws_size,
                              hipStream_t stream) {
  const float* s   = (const float*)d_in[0];
  const float* v   = (const float*)d_in[1];
  const float* re1 = (const float*)d_in[2];
  const float* re2 = (const float*)d_in[3];
  const float* f1  = (const float*)d_in[4];
  const float* f2  = (const float*)d_in[5];
  const float* u1  = (const float*)d_in[6];
  const float* u2  = (const float*)d_in[7];
  const int*  eidx = (const int*)d_in[8];
  const float* W1  = (const float*)d_in[9];
  const float* b1  = (const float*)d_in[10];
  const float* W2  = (const float*)d_in[11];
  const float* b2  = (const float*)d_in[12];
  const float* Wr  = (const float*)d_in[13];
  const float* br  = (const float*)d_in[14];

  int N = in_sizes[0] / F128;   // 10000
  int E = in_sizes[4];          // 320000

  // ws layout: sp | hp | w1p | w2p | bp | phi  (all 256B aligned)
  char* ws = (char*)d_ws;
  size_t off = 0;
  unsigned short* sp  = (unsigned short*)(ws + off);
  off += (size_t)N * F128 * 2;        off = (off + 255) & ~(size_t)255;
  unsigned short* hp  = (unsigned short*)(ws + off);
  off += (size_t)N * F128 * 2;        off = (off + 255) & ~(size_t)255;
  unsigned short* w1p = (unsigned short*)(ws + off);
  off += (size_t)8 * 4 * 512 * 2;     off = (off + 255) & ~(size_t)255;
  unsigned short* w2p = (unsigned short*)(ws + off);
  off += (size_t)48 * 4 * 512 * 2;    off = (off + 255) & ~(size_t)255;
  unsigned short* bp  = (unsigned short*)(ws + off);
  off += (size_t)48 * 2 * 512 * 2;    off = (off + 255) & ~(size_t)255;
  float* phi = (float*)(ws + off);

  float* outS = (float*)d_out;
  float* outV = outS + (size_t)N * F128;

  int totalInit = N * F128 * 4;
  init_out_kernel<<<(totalInit + 255) / 256, 256, 0, stream>>>(s, v, (float*)d_out, N);
  pack_s_kernel<<<(N * F128 + 255) / 256, 256, 0, stream>>>(s, sp, N);
  pack_w_kernel<<<(8 * 4 * 512 + 255) / 256, 256, 0, stream>>>(W1, w1p, 8, 4, F128);
  pack_w_kernel<<<(48 * 4 * 512 + 255) / 256, 256, 0, stream>>>(W2, w2p, 48, 4, SIXF);
  pack_bpad_kernel<<<(48 * 2 * 512 + 255) / 256, 256, 0, stream>>>(Wr, bp);
  node_hidden_kernel<<<N / 16, 256, 0, stream>>>(sp, w1p, b1, hp, N);
  node_phi_kernel<<<N / 16, 256, 0, stream>>>(hp, w2p, b2, phi, N);
  edge_kernel<<<E / 16, 256, 0, stream>>>(re1, re2, f1, f2, u1, u2, eidx,
                                          phi, v, bp, br, outS, outV, E);
}
